// MultiHeadSelfAttention_31129922961563
// MI455X (gfx1250) — matrix operand order repaired and checked by reading, not compiled, not run on hardware
//
#include <hip/hip_runtime.h>

typedef __bf16 bf16_t;
typedef __bf16 v16bf __attribute__((ext_vector_type(16)));
typedef float v8f __attribute__((ext_vector_type(8)));

#define N_EMBD 1024
#define N_HEADS 16
#define HEAD_DIM 64
#define B_SZ 4
#define T_SZ 2048
#define QKV_F (3 * N_EMBD)

union FragBF {
  v16bf v;
  unsigned int u[8];
  uint4 q[2];
};

// A-matrix 16x32 bf16 fragment, contiguous k order (as the K B-frags), strided-gather form (Q from global,
// once per block): lane M = lane&15, khalf = lane>>4,
// element pair v -> k = kbase + khalf*16 + v*2
__device__ __forceinline__ v16bf load_a_frag_gather(const bf16_t* rowp, int khalf, int kbase) {
  FragBF f;
#pragma unroll
  for (int i = 0; i < 8; ++i) {
    int k = kbase + khalf * 16 + (i << 1);
    f.u[i] = *(const unsigned int*)(rowp + k);
  }
  return f.v;
}

// 16 contiguous bf16 -> fragment (two b128 loads). Used for B-fragments
// (element i -> K = khalf*16 + i) and A/P tiles stored in fragment order.
__device__ __forceinline__ v16bf load_frag_contig(const bf16_t* rowp, int kbase) {
  FragBF f;
  f.q[0] = *(const uint4*)(rowp + kbase);
  f.q[1] = *(const uint4*)(rowp + kbase + 8);
  return f.v;
}

__device__ __forceinline__ v8f wmma_bf16(v16bf a, v16bf b, v8f c) {
  return __builtin_amdgcn_wmma_f32_16x16x32_bf16(false, a, false, b, (short)0, c,
                                                 false, false);
}

// gfx1250 async DMA: global memory -> LDS, per-lane 16B, tracked by ASYNCcnt.
// LDS byte offset = low 32 bits of the flat shared address (aperture layout).
__device__ __forceinline__ void async_ld_b128(void* lds_ptr, const void* gptr) {
  unsigned loff = (unsigned)(size_t)lds_ptr;
  asm volatile("global_load_async_to_lds_b128 %0, %1, off"
               :: "v"(loff), "v"(gptr) : "memory");
}
__device__ __forceinline__ void wait_async0() {
  asm volatile("s_wait_asynccnt 0" ::: "memory");
}

// Butterfly XOR exchange within 16-lane rows via v_permlane16_b32 (VALU, no LDS)
__device__ __forceinline__ float permx(float x, unsigned lo, unsigned hi) {
  union { float f; unsigned u; } a, r;
  a.f = x;
  r.u = __builtin_amdgcn_permlane16(a.u, a.u, lo, hi, false, false);
  return r.f;
}
__device__ __forceinline__ float rowmax16(float x) {
  x = fmaxf(x, permx(x, 0x67452301u, 0xEFCDAB89u));  // xor 1
  x = fmaxf(x, permx(x, 0x54761032u, 0xDCFE98BAu));  // xor 2
  x = fmaxf(x, permx(x, 0x32107654u, 0xBA98FEDCu));  // xor 4
  x = fmaxf(x, permx(x, 0xFEDCBA98u, 0x76543210u));  // xor 8
  return x;
}
__device__ __forceinline__ float rowsum16(float x) {
  x += permx(x, 0x67452301u, 0xEFCDAB89u);
  x += permx(x, 0x54761032u, 0xDCFE98BAu);
  x += permx(x, 0x32107654u, 0xBA98FEDCu);
  x += permx(x, 0xFEDCBA98u, 0x76543210u);
  return x;
}

__global__ __launch_bounds__(256) void cvt_f32_bf16(const float* __restrict__ in,
                                                    unsigned short* __restrict__ out,
                                                    int n4) {
  int i = blockIdx.x * 256 + threadIdx.x;
  if (i >= n4) return;
  float4 f = ((const float4*)in)[i];
  union { bf16_t b[4]; uint2 u; } o;
  o.b[0] = (bf16_t)f.x;
  o.b[1] = (bf16_t)f.y;
  o.b[2] = (bf16_t)f.z;
  o.b[3] = (bf16_t)f.w;
  ((uint2*)out)[i] = o.u;
}

// C[M,N] = A[M,K] * W[N,K]^T, bf16 inputs, f32 accumulate.
// 128x128 block tile, BK=32, 8 waves of 64x32. Double-buffered LDS filled by
// global_load_async_to_lds_b128 (one barrier per K-step). A tile stored in
// WMMA fragment order: chunk kk {0,8,16,24} -> row pos {0,16,8,24}.
// Columns < scaleCols get * 0.125 (folds attention 1/sqrt(d) into Q).
template <typename OutT>
__global__ __launch_bounds__(256) void gemm_bf16_nt(const unsigned short* __restrict__ Au,
                                                    const unsigned short* __restrict__ Wu,
                                                    OutT* __restrict__ C,
                                                    int M, int N, int K, int scaleCols) {
  const bf16_t* A = reinterpret_cast<const bf16_t*>(Au);
  const bf16_t* W = reinterpret_cast<const bf16_t*>(Wu);
  const int BK = 32, SK = 40;  // row stride 40 bf16 = 20 dwords: conflict-free, 16B-aligned
  __shared__ bf16_t sA[2 * 128 * SK];
  __shared__ bf16_t sB[2 * 128 * SK];
  const int tid = threadIdx.x;
  const int lane = tid & 31, wave = tid >> 5;
  const int n = lane & 15, khalf = lane >> 4;
  const int rowBase = blockIdx.y * 128, colBase = blockIdx.x * 128;
  const int waveM = (wave >> 2) * 64, waveN = (wave & 3) * 32;

  // cooperative tile-load coordinates: 2 passes x (16B of 8 bf16) per thread
  const int r0 = tid >> 2;
  const int kk = (tid & 3) * 8;
  const int apos = ((kk & 8) << 1) | ((kk & 16) >> 1);  // fragment-order position

  v8f acc[4][2];
#pragma unroll
  for (int i = 0; i < 4; ++i)
#pragma unroll
    for (int j = 0; j < 2; ++j)
#pragma unroll
      for (int r = 0; r < 8; ++r) acc[i][j][r] = 0.0f;

  auto issueTile = [&](int k0, int buf) {
    bf16_t* dA = sA + buf * (128 * SK);
    bf16_t* dB = sB + buf * (128 * SK);
#pragma unroll
    for (int c = 0; c < 2; ++c) {
      int r = c * 64 + r0;
      async_ld_b128(dA + r * SK + apos, A + (size_t)(rowBase + r) * K + k0 + kk);
      async_ld_b128(dB + r * SK + apos, W + (size_t)(colBase + r) * K + k0 + kk);
    }
  };
  issueTile(0, 0);

  const int nTiles = K / BK;
  for (int it = 0; it < nTiles; ++it) {
    wait_async0();     // this wave's async fills of the current buffer are done
    __syncthreads();   // ... and so are every other wave's
    if (it + 1 < nTiles) issueTile((it + 1) * BK, (it + 1) & 1);
    if (it + 2 < nTiles) {  // warm L2 one tile further ahead
      __builtin_prefetch(A + (size_t)(rowBase + r0) * K + (it + 2) * BK + kk, 0, 1);
      __builtin_prefetch(W + (size_t)(colBase + r0) * K + (it + 2) * BK + kk, 0, 1);
    }
    const bf16_t* cA = sA + (it & 1) * (128 * SK);
    const bf16_t* cB = sB + (it & 1) * (128 * SK);

    v16bf af[4], bfg[2];
#pragma unroll
    for (int i = 0; i < 4; ++i)
      af[i] = load_frag_contig(cA + (waveM + i * 16 + n) * SK, khalf * 16);
#pragma unroll
    for (int j = 0; j < 2; ++j)
      bfg[j] = load_frag_contig(cB + (waveN + j * 16 + n) * SK, khalf * 16);
#pragma unroll
    for (int i = 0; i < 4; ++i)
#pragma unroll
      for (int j = 0; j < 2; ++j) acc[i][j] = wmma_bf16(af[i], bfg[j], acc[i][j]);
  }

  const int mh = khalf;
#pragma unroll
  for (int i = 0; i < 4; ++i)
#pragma unroll
    for (int j = 0; j < 2; ++j) {
      int col = colBase + waveN + j * 16 + n;
#pragma unroll
      for (int r = 0; r < 8; ++r) {
        int row = rowBase + waveM + i * 16 + mh * 8 + r;
        float v = acc[i][j][r];
        v = (col < scaleCols) ? v * 0.125f : v;
        if constexpr (sizeof(OutT) == 4) {
          C[(size_t)row * N + col] = v;
        } else {
          union { bf16_t b; unsigned short s; } cv;
          cv.b = (bf16_t)v;
          C[(size_t)row * N + col] = cv.s;
        }
      }
    }
}

// Flash attention, causal; Q pre-scaled by 1/sqrt(d) in the QKV GEMM.
// Grid: (T/128, B*H). 8 waves x 16 queries. Uniform key loop; K tile via
// async global->LDS DMA, V tile via register transpose-scatter; double
// buffers, one barrier per key block. All K/V fragments preloaded so DS
// latency hides under S-WMMA + softmax.
__global__ __launch_bounds__(256) void flash_attn(const unsigned short* __restrict__ qkvu,
                                                  unsigned short* __restrict__ yu) {
  const bf16_t* qkv = reinterpret_cast<const bf16_t*>(qkvu);
  const int KSTR = 72;  // 32 keys x 64 d: 144B rows, 16B aligned, bank-stride 36
  const int VSTR = 40;  // V^T: 64 d rows x 32 keys, 80B rows
  const int PSTR = 40;  // per-wave P scratch 16x32, fragment order
  __shared__ bf16_t sK[2 * 32 * KSTR];
  __shared__ bf16_t sV[2 * 64 * VSTR];  // transposed: [d][key]
  __shared__ bf16_t sP[8 * 16 * PSTR];  // per-wave
  const int tid = threadIdx.x, lane = tid & 31, wave = tid >> 5;
  const int n = lane & 15, mh = lane >> 4;
  const int qb = blockIdx.x;
  const int bh = blockIdx.y;
  const int b = bh >> 4, h = bh & 15;
  const int q0 = qb * 128;
  const int qbase = q0 + wave * 16;
  const size_t basebt = (size_t)(b * T_SZ) * QKV_F + h * HEAD_DIM;

  // Q fragments (A-layout, gathered from global once, resident all iterations)
  const bf16_t* qrow = qkv + basebt + (size_t)(qbase + n) * QKV_F;
  v16bf aQ0 = load_a_frag_gather(qrow, mh, 0);
  v16bf aQ1 = load_a_frag_gather(qrow, mh, 32);

  float mrow[8], lrow[8];
  v8f o0, o1, o2, o3;
#pragma unroll
  for (int r = 0; r < 8; ++r) {
    mrow[r] = -__builtin_inff();
    lrow[r] = 0.0f;
    o0[r] = 0.0f; o1[r] = 0.0f; o2[r] = 0.0f; o3[r] = 0.0f;
  }

  bf16_t* sPw = sP + wave * 16 * PSTR;
  // P fragment-order column position for key n / key 16+n
  const int pos0 = (n & 7) + ((n & 8) << 1);

  // cooperative K/V staging: 256 threads x 8 contiguous d each
  const int key = tid >> 3, d0 = (tid & 7) * 8;
  uint4 vreg;
  auto issueK = [&](int j, int buf) {
    async_ld_b128(sK + buf * (32 * KSTR) + key * KSTR + d0,
                  qkv + basebt + (size_t)(j + key) * QKV_F + N_EMBD + d0);
  };
  auto loadV = [&](int j) {
    vreg = *(const uint4*)(qkv + basebt + (size_t)(j + key) * QKV_F + 2 * N_EMBD + d0);
  };
  issueK(0, 0);
  loadV(0);

  const int nkb = qb * 4 + 4;  // key blocks of 32 up to this block's causal bound
  for (int kb = 0; kb < nkb; ++kb) {
    const int j = kb * 32;
    const int buf = kb & 1;
    {  // scatter this tile's V (loaded last iteration) into its buffer
      bf16_t* dV = sV + buf * (64 * VSTR);
      union { uint4 q; bf16_t e[8]; } vv;
      vv.q = vreg;
#pragma unroll
      for (int e = 0; e < 8; ++e) dV[(d0 + e) * VSTR + ((key & 7) | ((key & 8) << 1) | ((key & 16) >> 1))] = vv.e[e];
    }
    wait_async0();     // this wave's K-tile DMA done
    __syncthreads();   // all waves' K DMA + V scatter visible
    if (kb + 1 < nkb) {
      issueK(j + 32, buf ^ 1);  // overlap next tile with compute
      loadV(j + 32);
    }
    const bf16_t* cK = sK + buf * (32 * KSTR);
    const bf16_t* cV = sV + buf * (64 * VSTR);

    // Preload ALL fragments for this tile so DS latency overlaps WMMA/softmax:
    v16bf kf[4], vf[4];
    kf[0] = load_frag_contig(cK + n * KSTR, mh * 16);
    kf[1] = load_frag_contig(cK + n * KSTR, 32 + mh * 16);
    kf[2] = load_frag_contig(cK + (16 + n) * KSTR, mh * 16);
    kf[3] = load_frag_contig(cK + (16 + n) * KSTR, 32 + mh * 16);
#pragma unroll
    for (int c = 0; c < 4; ++c)
      vf[c] = load_frag_contig(cV + (c * 16 + n) * VSTR, mh * 16);

    // S = Q * K^T : two 16x16 key tiles, d=64 split into two K=32 wmma
    v8f s0, s1;
#pragma unroll
    for (int r = 0; r < 8; ++r) { s0[r] = 0.0f; s1[r] = 0.0f; }
    s0 = wmma_bf16(aQ0, kf[0], s0);
    s0 = wmma_bf16(aQ1, kf[1], s0);
    s1 = wmma_bf16(aQ0, kf[2], s1);
    s1 = wmma_bf16(aQ1, kf[3], s1);

    const int k0i = j + n, k1i = j + 16 + n;
#pragma unroll
    for (int r = 0; r < 8; ++r) {
      const int qr = qbase + mh * 8 + r;
      float a = (k0i <= qr) ? s0[r] : -__builtin_inff();
      float c = (k1i <= qr) ? s1[r] : -__builtin_inff();
      const float mx = rowmax16(fmaxf(a, c));
      const float mnew = fmaxf(mrow[r], mx);
      const float alpha = __expf(mrow[r] - mnew);
      mrow[r] = mnew;
      const float p0 = __expf(a - mnew);
      const float p1 = __expf(c - mnew);
      const float ps = rowsum16(p0 + p1);
      lrow[r] = lrow[r] * alpha + ps;
      o0[r] *= alpha; o1[r] *= alpha; o2[r] *= alpha; o3[r] *= alpha;
      sPw[(mh * 8 + r) * PSTR + pos0] = (bf16_t)p0;      // transpose P via LDS,
      sPw[(mh * 8 + r) * PSTR + pos0 + 8] = (bf16_t)p1;  // fragment-order columns
    }
    // per-wave LDS RAW: DS ops are in-order within a wave; fence the compiler
    asm volatile("" ::: "memory");

    // O += P(16x32) * V(32x64)
    v16bf aP = load_frag_contig(sPw + n * PSTR, mh * 16);
    o0 = wmma_bf16(aP, vf[0], o0);
    o1 = wmma_bf16(aP, vf[1], o1);
    o2 = wmma_bf16(aP, vf[2], o2);
    o3 = wmma_bf16(aP, vf[3], o3);
  }

  // normalize and store y[b,t,h,d] as bf16 rows of [8192,1024]
#pragma unroll
  for (int r = 0; r < 8; ++r) {
    const float inv = 1.0f / lrow[r];
    const int qr = qbase + mh * 8 + r;
    unsigned short* yp = yu + (size_t)(b * T_SZ + qr) * N_EMBD + h * HEAD_DIM;
    union { bf16_t bb; unsigned short s; } cv;
    cv.bb = (bf16_t)(o0[r] * inv); yp[n] = cv.s;
    cv.bb = (bf16_t)(o1[r] * inv); yp[16 + n] = cv.s;
    cv.bb = (bf16_t)(o2[r] * inv); yp[32 + n] = cv.s;
    cv.bb = (bf16_t)(o3[r] * inv); yp[48 + n] = cv.s;
  }
}

extern "C" void kernel_launch(void* const* d_in, const int* in_sizes, int n_in,
                              void* d_out, int out_size, void* d_ws, size_t ws_size,
                              hipStream_t stream) {
  (void)in_sizes; (void)n_in; (void)out_size; (void)ws_size;
  const float* x = (const float*)d_in[0];
  const float* w_qkv = (const float*)d_in[1];
  const float* w_out = (const float*)d_in[2];
  float* out = (float*)d_out;

  const int M = B_SZ * T_SZ;  // 8192
  char* ws = (char*)d_ws;
  size_t off = 0;
  auto alloc = [&](size_t bytes) {
    void* p = ws + off;
    off = (off + bytes + 255) & ~(size_t)255;
    return p;
  };
  unsigned short* xb = (unsigned short*)alloc((size_t)M * N_EMBD * 2);        // 16 MB
  unsigned short* wqb = (unsigned short*)alloc((size_t)QKV_F * N_EMBD * 2);   // 6 MB
  unsigned short* wob = (unsigned short*)alloc((size_t)N_EMBD * N_EMBD * 2);  // 2 MB
  unsigned short* qkvb = (unsigned short*)alloc((size_t)M * QKV_F * 2);       // 48 MB
  unsigned short* yb = (unsigned short*)alloc((size_t)M * N_EMBD * 2);        // 16 MB

  cvt_f32_bf16<<<(M * N_EMBD / 4 + 255) / 256, 256, 0, stream>>>(x, xb, M * N_EMBD / 4);
  cvt_f32_bf16<<<(QKV_F * N_EMBD / 4 + 255) / 256, 256, 0, stream>>>(w_qkv, wqb,
                                                                     QKV_F * N_EMBD / 4);
  cvt_f32_bf16<<<(N_EMBD * N_EMBD / 4 + 255) / 256, 256, 0, stream>>>(w_out, wob,
                                                                      N_EMBD * N_EMBD / 4);

  // QKV projection; scale Q columns (f < 1024) by 1/sqrt(HEAD_DIM)
  dim3 g1(QKV_F / 128, M / 128);
  gemm_bf16_nt<unsigned short><<<g1, 256, 0, stream>>>(xb, wqb, qkvb, M, QKV_F, N_EMBD,
                                                       N_EMBD);

  dim3 g2(T_SZ / 128, B_SZ * N_HEADS);
  flash_attn<<<g2, 256, 0, stream>>>(qkvb, yb);

  dim3 g3(N_EMBD / 128, M / 128);
  gemm_bf16_nt<float><<<g3, 256, 0, stream>>>(yb, wob, out, M, N_EMBD, N_EMBD, 0);
}
